// _SimpleNewMFModel_72292889526855
// MI455X (gfx1250) — matrix operand order repaired and checked by reading, compiled, not run on hardware
//
#include <hip/hip_runtime.h>
#include <stdint.h>

// ---------------------------------------------------------------------------
// SimpleNewMF fused kernel for gfx1250 (MI455X).
// Four 1024x50000x64 GEMMs done with v_wmma_f32_16x16x32_bf16, fully fused
// with the Pearson scaling, L1 / sum / dot reductions and min/max rescaling,
// so no [B, 50000] intermediate is ever materialized.
// ---------------------------------------------------------------------------

typedef __attribute__((ext_vector_type(16))) __bf16 bf16x16;
typedef __attribute__((ext_vector_type(8)))  __bf16 bf16x8;
typedef __attribute__((ext_vector_type(8)))  float  f32x8;
typedef __attribute__((ext_vector_type(4)))  unsigned int u32x4;

#define B_SZ   1024
#define DIM    64
#define NROWS  50000
#define NTILES (NROWS / 16)     // 3125 (exact)
#define TPC    125              // i-tiles per chunk
#define NCHUNK (NTILES / TPC)   // 25 (exact)
#define EPS    1e-12f

union B128 { u32x4 u; bf16x8 h; };

__device__ __forceinline__ unsigned short f2bf_rne(float f) {
  unsigned u = __float_as_uint(f);
  u += 0x7FFFu + ((u >> 16) & 1u);          // round to nearest even
  return (unsigned short)(u >> 16);
}
// monotone float -> uint map (order preserving) for atomicMin/Max
__device__ __forceinline__ unsigned mapf(float x) {
  unsigned b = __float_as_uint(x);
  return (b & 0x80000000u) ? ~b : (b | 0x80000000u);
}
__device__ __forceinline__ float unmapf(unsigned u) {
  unsigned b = (u & 0x80000000u) ? (u ^ 0x80000000u) : ~u;
  return __uint_as_float(b);
}
__device__ __forceinline__ float wred_add32(float v) {   // full wave32 reduce
  v += __shfl_xor(v, 16, 32);
  v += __shfl_xor(v,  8, 32);
  v += __shfl_xor(v,  4, 32);
  v += __shfl_xor(v,  2, 32);
  v += __shfl_xor(v,  1, 32);
  return v;
}

// A operand (16x32 bf16): lane m=lane&15, kbase=8*(lane>=16);
// VGPR0-3 = elems [k0+kbase .. +7], VGPR4-7 = elems [k0+16+kbase .. +7]
__device__ __forceinline__ bf16x16 loadA(const unsigned short* row, int k0, int kb) {
  B128 lo, hi;
  lo.u = *(const u32x4*)(row + k0 + kb);
  hi.u = *(const u32x4*)(row + k0 + 16 + kb);
  return __builtin_shufflevector(lo.h, hi.h, 0,1,2,3,4,5,6,7,8,9,10,11,12,13,14,15);
}
// B operand (32x16 bf16): lane n=lane&15, elems [koff .. +7] and [koff+16 .. +7], koff includes 8*(lane>=16)
__device__ __forceinline__ bf16x16 loadB(const unsigned short* row, int koff) {
  B128 lo, hi;
  lo.u = *(const u32x4*)(row + koff);
  hi.u = *(const u32x4*)(row + koff + 16);
  return __builtin_shufflevector(lo.h, hi.h, 0,1,2,3,4,5,6,7,8,9,10,11,12,13,14,15);
}

// --------------------------------------------------------------------------
// Kernel 0: init accumulators
// --------------------------------------------------------------------------
__global__ __launch_bounds__(256) void init_acc(
    float* uS, float* uL1, float* uD, float* iS, float* iL1, float* iD,
    unsigned* uMn, unsigned* uMx, unsigned* iMn, unsigned* iMx) {
  int t = blockIdx.x * blockDim.x + threadIdx.x;
  if (t >= B_SZ) return;
  uS[t] = 0.f; uL1[t] = 0.f; uD[t] = 0.f;
  iS[t] = 0.f; iL1[t] = 0.f; iD[t] = 0.f;
  uMn[t] = 0xFFFFFFFFu; uMx[t] = 0u;
  iMn[t] = 0xFFFFFFFFu; iMx[t] = 0u;
}

// --------------------------------------------------------------------------
// Kernel 1: per-row prep of the big tables.
// kind 0/1: mean-center W_u / W_i rows, emit bf16 + rsqrt(sumsq)
// kind 2/3: plain bf16 convert of W_u_vi / W_i_uj
// One wave per row.
// --------------------------------------------------------------------------
__global__ __launch_bounds__(256) void prep_rows(
    const float* __restrict__ Wu, const float* __restrict__ Wi,
    const float* __restrict__ Wuvi, const float* __restrict__ Wiuj,
    unsigned short* __restrict__ WuC, unsigned short* __restrict__ WiC,
    unsigned short* __restrict__ WuviB, unsigned short* __restrict__ WiujB,
    float* __restrict__ rsBu, float* __restrict__ rsBi) {
  int wid  = (blockIdx.x * blockDim.x + threadIdx.x) >> 5;
  int lane = threadIdx.x & 31;
  if (wid >= 4 * NROWS) return;
  int kind = wid / NROWS;
  int row  = wid % NROWS;
  const float* src; unsigned short* dst; float* rs = nullptr;
  if      (kind == 0) { src = Wu   + (size_t)row * DIM; dst = WuC   + (size_t)row * DIM; rs = rsBu; }
  else if (kind == 1) { src = Wi   + (size_t)row * DIM; dst = WiC   + (size_t)row * DIM; rs = rsBi; }
  else if (kind == 2) { src = Wuvi + (size_t)row * DIM; dst = WuviB + (size_t)row * DIM; }
  else                { src = Wiuj + (size_t)row * DIM; dst = WiujB + (size_t)row * DIM; }
  float x0 = src[lane], x1 = src[lane + 32];
  if (kind < 2) {
    float mean = wred_add32(x0 + x1) * (1.0f / 64.0f);
    x0 -= mean; x1 -= mean;
    float ss = wred_add32(x0 * x0 + x1 * x1);
    if (lane == 0) rs[row] = rsqrtf(ss);
  }
  dst[lane]      = f2bf_rne(x0);
  dst[lane + 32] = f2bf_rne(x1);
}

// --------------------------------------------------------------------------
// Kernel 2: gather+prep batch rows (one wave per b)
// --------------------------------------------------------------------------
__device__ __forceinline__ void center_row(const float* src, unsigned short* dst,
                                           float* rsOut, int lane) {
  float x0 = src[lane], x1 = src[lane + 32];
  float mean = wred_add32(x0 + x1) * (1.0f / 64.0f);
  x0 -= mean; x1 -= mean;
  float ss = wred_add32(x0 * x0 + x1 * x1);
  if (lane == 0) *rsOut = rsqrtf(ss);
  dst[lane]      = f2bf_rne(x0);
  dst[lane + 32] = f2bf_rne(x1);
}
__device__ __forceinline__ void conv_row(const float* src, unsigned short* dst, int lane) {
  dst[lane]      = f2bf_rne(src[lane]);
  dst[lane + 32] = f2bf_rne(src[lane + 32]);
}

__global__ __launch_bounds__(256) void prep_batch(
    const int* __restrict__ user, const int* __restrict__ item,
    const float* __restrict__ Wu, const float* __restrict__ Wi,
    const float* __restrict__ Wivi, const float* __restrict__ Wuuj,
    unsigned short* __restrict__ euc, unsigned short* __restrict__ eic,
    unsigned short* __restrict__ evi, unsigned short* __restrict__ euj,
    float* __restrict__ rsAu, float* __restrict__ rsAi) {
  int wid  = (blockIdx.x * blockDim.x + threadIdx.x) >> 5;
  int lane = threadIdx.x & 31;
  if (wid >= B_SZ) return;
  int u = user[wid], it = item[wid];
  center_row(Wu + (size_t)u  * DIM, euc + (size_t)wid * DIM, rsAu + wid, lane);
  center_row(Wi + (size_t)it * DIM, eic + (size_t)wid * DIM, rsAi + wid, lane);
  conv_row(Wivi + (size_t)it * DIM, evi + (size_t)wid * DIM, lane);
  conv_row(Wuuj + (size_t)u  * DIM, euj + (size_t)wid * DIM, lane);
}

// --------------------------------------------------------------------------
// Kernel 3: the fused dual-GEMM + reduction engine (used for both sides).
//  sim[m,n]   = (Asim[m] . Bsim[n]) * rsA[m] * rsB[n]   (Pearson)
//  alpha[m,n] =  Aalp[m] . Balp[n]
//  accumulate over n: S += sim, L1 += |sim|, D += sim*alpha, min/max alpha
// One wave handles one 16-row b-tile x TPC i-tiles.
// --------------------------------------------------------------------------
__global__ __launch_bounds__(256) void side_gemm(
    const unsigned short* __restrict__ Asim, const unsigned short* __restrict__ Aalp,
    const unsigned short* __restrict__ Bsim, const unsigned short* __restrict__ Balp,
    const float* __restrict__ rsA, const float* __restrict__ rsB,
    float* accS, float* accL1, float* accD, unsigned* accMn, unsigned* accMx) {
  int wid  = (blockIdx.x * blockDim.x + threadIdx.x) >> 5;
  int lane = threadIdx.x & 31;
  int btile = wid & 63;        // 64 b-tiles of 16 rows
  int chunk = wid >> 6;
  int t0 = chunk * TPC;
  if (t0 >= NTILES) return;
  int t1 = t0 + TPC;           // exact: 3125 = 25 * 125

  int m16 = lane & 15;
  int hi  = (lane >> 4) & 1;
  int kb  = hi * 8;            // A K-base
  int kh  = hi * 8;            // B K-base

  const unsigned short* arowS = Asim + (size_t)(btile * 16 + m16) * DIM;
  const unsigned short* arowA = Aalp + (size_t)(btile * 16 + m16) * DIM;
  bf16x16 aS0 = loadA(arowS, 0, kb), aS1 = loadA(arowS, 32, kb);
  bf16x16 aA0 = loadA(arowA, 0, kb), aA1 = loadA(arowA, 32, kb);

  float rsAv[8];
#pragma unroll
  for (int r = 0; r < 8; ++r) rsAv[r] = rsA[btile * 16 + r + hi * 8];

  float sS[8], sL[8], sD[8], mn[8], mx[8];
#pragma unroll
  for (int r = 0; r < 8; ++r) {
    sS[r] = 0.f; sL[r] = 0.f; sD[r] = 0.f;
    mn[r] = __builtin_inff(); mx[r] = -__builtin_inff();
  }

  for (int t = t0; t < t1; ++t) {
    const unsigned short* browS = Bsim + (size_t)(t * 16 + m16) * DIM;
    const unsigned short* browA = Balp + (size_t)(t * 16 + m16) * DIM;
    bf16x16 bS0 = loadB(browS, kh), bS1 = loadB(browS, 32 + kh);
    bf16x16 bA0 = loadB(browA, kh), bA1 = loadB(browA, 32 + kh);
    float rb = rsB[t * 16 + m16];

    f32x8 cS = {};
    cS = __builtin_amdgcn_wmma_f32_16x16x32_bf16(false, aS0, false, bS0, (short)0, cS, false, false);
    cS = __builtin_amdgcn_wmma_f32_16x16x32_bf16(false, aS1, false, bS1, (short)0, cS, false, false);
    f32x8 cA = {};
    cA = __builtin_amdgcn_wmma_f32_16x16x32_bf16(false, aA0, false, bA0, (short)0, cA, false, false);
    cA = __builtin_amdgcn_wmma_f32_16x16x32_bf16(false, aA1, false, bA1, (short)0, cA, false, false);

#pragma unroll
    for (int r = 0; r < 8; ++r) {
      float sim = cS[r] * rsAv[r] * rb;
      float a   = cA[r];
      sS[r] += sim;
      sL[r] += fabsf(sim);
      sD[r] += sim * a;
      mn[r] = fminf(mn[r], a);
      mx[r] = fmaxf(mx[r], a);
    }
  }

  // reduce over n (16 lanes of each half), then merge globally
#pragma unroll
  for (int r = 0; r < 8; ++r) {
    float vS = sS[r], vL = sL[r], vD = sD[r], vm = mn[r], vM = mx[r];
#pragma unroll
    for (int msk = 8; msk >= 1; msk >>= 1) {
      vS += __shfl_xor(vS, msk, 32);
      vL += __shfl_xor(vL, msk, 32);
      vD += __shfl_xor(vD, msk, 32);
      vm = fminf(vm, __shfl_xor(vm, msk, 32));
      vM = fmaxf(vM, __shfl_xor(vM, msk, 32));
    }
    if (m16 == 0) {
      int m = btile * 16 + r + hi * 8;
      atomicAdd(&accS[m],  vS);
      atomicAdd(&accL1[m], vL);
      atomicAdd(&accD[m],  vD);
      atomicMin(&accMn[m], mapf(vm));
      atomicMax(&accMx[m], mapf(vM));
    }
  }
}

// --------------------------------------------------------------------------
// Kernel 4: finalize. base dot + analytic self-term subtraction + combine.
// --------------------------------------------------------------------------
__device__ __forceinline__ float dot64(const float* a, const float* b, int lane) {
  float s = a[lane] * b[lane] + a[lane + 32] * b[lane + 32];
  return wred_add32(s);
}

__global__ __launch_bounds__(256) void finalize(
    const int* __restrict__ user, const int* __restrict__ item,
    const float* __restrict__ Wu, const float* __restrict__ Wi,
    const float* __restrict__ Wuvi, const float* __restrict__ Wivi,
    const float* __restrict__ Wuuj, const float* __restrict__ Wiuj,
    const float* uS, const float* uL1, const float* uD,
    const unsigned* uMn, const unsigned* uMx,
    const float* iS, const float* iL1, const float* iD,
    const unsigned* iMn, const unsigned* iMx,
    float* __restrict__ out) {
  int wid  = (blockIdx.x * blockDim.x + threadIdx.x) >> 5;
  int lane = threadIdx.x & 31;
  if (wid >= B_SZ) return;
  int u = user[wid], it = item[wid];
  float base = dot64(Wu   + (size_t)u * DIM, Wi   + (size_t)it * DIM, lane);
  float asu  = dot64(Wuvi + (size_t)u * DIM, Wivi + (size_t)it * DIM, lane); // alpha_vi self
  float asi  = dot64(Wuuj + (size_t)u * DIM, Wiuj + (size_t)it * DIM, lane); // alpha_uj self
  if (lane == 0) {
    // self sim == corr(row,row) == 1 exactly in reference semantics
    float S  = uS[wid] - 1.0f;
    float L  = fmaxf(uL1[wid] - 1.0f, EPS);
    float D  = uD[wid] - asu;
    float mnv = unmapf(uMn[wid]), mxv = unmapf(uMx[wid]);
    float termU = (D - mnv * S) / ((mxv - mnv) * L);

    S  = iS[wid] - 1.0f;
    L  = fmaxf(iL1[wid] - 1.0f, EPS);
    D  = iD[wid] - asi;
    mnv = unmapf(iMn[wid]); mxv = unmapf(iMx[wid]);
    float termI = (D - mnv * S) / ((mxv - mnv) * L);

    out[wid] = base + termU + termI;
  }
}

// --------------------------------------------------------------------------
// Host launcher
// --------------------------------------------------------------------------
extern "C" void kernel_launch(void* const* d_in, const int* in_sizes, int n_in,
                              void* d_out, int out_size, void* d_ws, size_t ws_size,
                              hipStream_t stream) {
  const int*   user = (const int*)d_in[0];
  const int*   item = (const int*)d_in[1];
  const float* Wu   = (const float*)d_in[2];
  const float* Wi   = (const float*)d_in[3];
  const float* Wuvi = (const float*)d_in[4];
  const float* Wivi = (const float*)d_in[5];
  const float* Wuuj = (const float*)d_in[6];
  const float* Wiuj = (const float*)d_in[7];
  float* out = (float*)d_out;

  // ---- carve workspace (all 256B aligned) ----
  char*  ws  = (char*)d_ws;
  size_t off = 0;
  auto carve = [&](size_t bytes) -> void* {
    off = (off + 255) & ~(size_t)255;
    void* p = ws + off;
    off += bytes;
    return p;
  };
  float*    accS_u  = (float*)carve(B_SZ * 4);
  float*    accL1_u = (float*)carve(B_SZ * 4);
  float*    accD_u  = (float*)carve(B_SZ * 4);
  float*    accS_i  = (float*)carve(B_SZ * 4);
  float*    accL1_i = (float*)carve(B_SZ * 4);
  float*    accD_i  = (float*)carve(B_SZ * 4);
  unsigned* mnU     = (unsigned*)carve(B_SZ * 4);
  unsigned* mxU     = (unsigned*)carve(B_SZ * 4);
  unsigned* mnI     = (unsigned*)carve(B_SZ * 4);
  unsigned* mxI     = (unsigned*)carve(B_SZ * 4);
  float*    rsAu    = (float*)carve(B_SZ * 4);
  float*    rsAi    = (float*)carve(B_SZ * 4);
  float*    rsBu    = (float*)carve((size_t)NROWS * 4);
  float*    rsBi    = (float*)carve((size_t)NROWS * 4);
  unsigned short* euc   = (unsigned short*)carve((size_t)B_SZ  * DIM * 2);
  unsigned short* eic   = (unsigned short*)carve((size_t)B_SZ  * DIM * 2);
  unsigned short* evi   = (unsigned short*)carve((size_t)B_SZ  * DIM * 2);
  unsigned short* euj   = (unsigned short*)carve((size_t)B_SZ  * DIM * 2);
  unsigned short* WuC   = (unsigned short*)carve((size_t)NROWS * DIM * 2);
  unsigned short* WiC   = (unsigned short*)carve((size_t)NROWS * DIM * 2);
  unsigned short* WuviB = (unsigned short*)carve((size_t)NROWS * DIM * 2);
  unsigned short* WiujB = (unsigned short*)carve((size_t)NROWS * DIM * 2);
  (void)ws_size; (void)in_sizes; (void)n_in; (void)out_size;

  // ---- pipeline ----
  init_acc<<<(B_SZ + 255) / 256, 256, 0, stream>>>(
      accS_u, accL1_u, accD_u, accS_i, accL1_i, accD_i, mnU, mxU, mnI, mxI);

  // 4*50000 rows, one wave each, 8 waves/block
  prep_rows<<<(4 * NROWS + 7) / 8, 256, 0, stream>>>(
      Wu, Wi, Wuvi, Wiuj, WuC, WiC, WuviB, WiujB, rsBu, rsBi);

  prep_batch<<<(B_SZ + 7) / 8, 256, 0, stream>>>(
      user, item, Wu, Wi, Wivi, Wuuj, euc, eic, evi, euj, rsAu, rsAi);

  // user side: sim = euc x WuC^T (Pearson), alpha = evi x WuviB^T
  side_gemm<<<(64 * NCHUNK) / 8, 256, 0, stream>>>(
      euc, evi, WuC, WuviB, rsAu, rsBu, accS_u, accL1_u, accD_u, mnU, mxU);
  // item side: sim = eic x WiC^T, alpha = euj x WiujB^T
  side_gemm<<<(64 * NCHUNK) / 8, 256, 0, stream>>>(
      eic, euj, WiC, WiujB, rsAi, rsBi, accS_i, accL1_i, accD_i, mnI, mxI);

  finalize<<<(B_SZ + 7) / 8, 256, 0, stream>>>(
      user, item, Wu, Wi, Wuvi, Wivi, Wuuj, Wiuj,
      accS_u, accL1_u, accD_u, mnU, mxU,
      accS_i, accL1_i, accD_i, mnI, mxI, out);
}
